// LocalGeometryRegularizer_73005854097893
// MI455X (gfx1250) — matrix operand order repaired and checked by reading, not compiled, not run on hardware
//
#include <hip/hip_runtime.h>

// ---------------------------------------------------------------------------
// LocalGeometryRegularizer for MI455X (gfx1250, wave32, WMMA).
// N=8192 points, D=512 dims, k=15 neighbors.
// Gram GEMM via v_wmma_f32_16x16x32_bf16 with bf16 hi/lo error split
// (3 WMMAs per 32-K chunk), fused with streaming per-row top-16 on squared
// distances (sqrt deferred to the tiny merge kernel; monotone => same order).
// Row-block = 32 halves L2 B-traffic; kc loop kept rolled (unroll 2) so
// A-fragments are re-read from LDS instead of being hoisted + spilled.
// ---------------------------------------------------------------------------

#define NPTS   8192
#define DIMS   512
#define KNN    15
#define NPAIR  105          // 15*14/2
#define EPSF   1e-8f
#define MBLK   32           // rows per workgroup in the knn kernel
#define PARTS  4            // partial top-16 lists per row (waves per row-group)
#define APAD   8            // LDS row pad -> stride 520 elems = 1040 B (16B aligned)
#define ANGWAVES 2          // waves per workgroup in the angular kernel

typedef __attribute__((ext_vector_type(16))) __bf16       v16bf;
typedef __attribute__((ext_vector_type(8)))  float        v8f;
typedef __attribute__((ext_vector_type(8)))  unsigned int v8u;

__device__ __forceinline__ unsigned short f2bf_rne(float x) {
  unsigned int u = __builtin_bit_cast(unsigned int, x);
  unsigned int r = u + 0x7FFFu + ((u >> 16) & 1u);
  return (unsigned short)(r >> 16);
}
__device__ __forceinline__ float bf2f(unsigned short h) {
  return __builtin_bit_cast(float, ((unsigned int)h) << 16);
}
__device__ __forceinline__ float wave_sum(float v) {
  #pragma unroll
  for (int off = 16; off > 0; off >>= 1) v += __shfl_xor(v, off, 32);
  return v;
}

// --- WMMA fragment loaders (contiguous K order, same for A and B) ------
// A (16x32 bf16): lane L holds row M=L&15. VGPR v: K = 2v+16*half
__device__ __forceinline__ v8u load_frag_a(const unsigned short* rowp, int kb, int half) {
  v8u u;
  #pragma unroll
  for (int v = 0; v < 8; ++v) {
    int koff = 2 * v + 16 * half;
    u[v] = *(const unsigned int*)(rowp + kb + koff);
  }
  return u;
}
// B (32x16 bf16): lane L holds col N=L&15 (a row of E). VGPR v: K = 2v+16*half
__device__ __forceinline__ v8u load_frag_b(const unsigned short* rowp, int kb, int half) {
  v8u u;
  #pragma unroll
  for (int v = 0; v < 8; ++v)
    u[v] = *(const unsigned int*)(rowp + kb + 2 * v + 16 * half);
  return u;
}
__device__ __forceinline__ v8f wmma_bf16(v8u a, v8u b, v8f c) {
  return __builtin_amdgcn_wmma_f32_16x16x32_bf16(
      /*neg_a=*/false, __builtin_bit_cast(v16bf, a),
      /*neg_b=*/false, __builtin_bit_cast(v16bf, b),
      /*c_mod=*/(short)0, c, /*reuse_a=*/false, /*reuse_b=*/false);
}

// ---------------------------------------------------------------------------
// Kernel 0: zero the scalar accumulators (harness poisons ws with 0xAA).
__global__ void lgr_init(float* acc) {
  if (threadIdx.x < 8) acc[threadIdx.x] = 0.f;
}

// ---------------------------------------------------------------------------
// Kernel 1: fp32 -> bf16 hi/lo split planes + fp32 row squared norms.
// grid (NPTS/8, 2), block 256 (8 waves, one row per wave).
__global__ void lgr_prep(const float* __restrict__ emb, const float* __restrict__ ref,
                         unsigned short* EhiE, unsigned short* EloE, float* sqE,
                         unsigned short* EhiR, unsigned short* EloR, float* sqR) {
  const float* src       = blockIdx.y ? ref  : emb;
  unsigned short* Ehi    = blockIdx.y ? EhiR : EhiE;
  unsigned short* Elo    = blockIdx.y ? EloR : EloE;
  float* sq              = blockIdx.y ? sqR  : sqE;
  const int w = threadIdx.x >> 5, lane = threadIdx.x & 31;
  const int row = blockIdx.x * 8 + w;
  float s = 0.f;
  #pragma unroll
  for (int t = 0; t < 16; ++t) {
    int c = lane + 32 * t;
    float x = src[row * DIMS + c];
    unsigned short h = f2bf_rne(x);
    unsigned short l = f2bf_rne(x - bf2f(h));
    Ehi[row * DIMS + c] = h;
    Elo[row * DIMS + c] = l;
    s += x * x;
  }
  s = wave_sum(s);
  if (lane == 0) sq[row] = s;
}

// ---------------------------------------------------------------------------
// Kernel 2: fused gram-GEMM + streaming per-row top-16 of SQUARED distances
// (excluding self). WG owns 32 rows (waves 0-3 -> rows 0-15, waves 4-7 ->
// rows 16-31); each quartet strides over the 512 column tiles.
// d2 = |a|^2+|b|^2-2*a.b, a.b from 3x bf16 WMMA (hi/lo split).
// grid NPTS/32, block 256.
__global__ void lgr_knn(const unsigned short* __restrict__ Ehi,
                        const unsigned short* __restrict__ Elo,
                        const float* __restrict__ sq,
                        float* __restrict__ pld, int* __restrict__ pli) {
  __shared__ unsigned short sAhi[MBLK][DIMS + APAD];
  __shared__ unsigned short sAlo[MBLK][DIMS + APAD];
  __shared__ float sSq[MBLK];
  __shared__ float sTile[8][16][16];
  __shared__ float sLd[8][16][16];
  __shared__ int   sLi[8][16][16];

  const int tid = threadIdx.x, w = tid >> 5, lane = tid & 31;
  const int half = lane >> 4, nn = lane & 15;
  const int rowGroup = w >> 2;          // 0: rows 0-15, 1: rows 16-31
  const int wsub = w & 3;               // column-tile stride phase
  const int rowBase = blockIdx.x * MBLK;
  const float FINF = __builtin_huge_valf();

  for (int i = tid; i < MBLK * DIMS; i += 256) {
    int r = i >> 9, c = i & (DIMS - 1);
    sAhi[r][c] = Ehi[(rowBase + r) * DIMS + c];
    sAlo[r][c] = Elo[(rowBase + r) * DIMS + c];
  }
  if (tid < MBLK) sSq[tid] = sq[rowBase + tid];
  __syncthreads();

  for (int e = lane; e < 256; e += 32) {
    sLd[w][e >> 4][e & 15] = FINF;
    sLi[w][e >> 4][e & 15] = -1;
  }
  float rowmax = FINF;                          // meaningful for lanes 0..15
  const int gi = rowBase + rowGroup * 16 + lane; // row owned by lane (lane<16)

  const unsigned short* ahip = &sAhi[rowGroup * 16 + nn][0];
  const unsigned short* alop = &sAlo[rowGroup * 16 + nn][0];

  #pragma unroll 1
  for (int tile = wsub; tile < NPTS / 16; tile += PARTS) {
    const int colBase = tile * 16;
    const int n = colBase + nn;
    const unsigned short* bhip = Ehi + n * DIMS;
    const unsigned short* blop = Elo + n * DIMS;
    if (tile + PARTS < NPTS / 16) {            // prefetch next B tile rows
      __builtin_prefetch(bhip + PARTS * 16 * DIMS, 0, 1);
      __builtin_prefetch(blop + PARTS * 16 * DIMS, 0, 1);
    }
    v8f C = {};
    #pragma unroll 2
    for (int kc = 0; kc < DIMS / 32; ++kc) {
      int kb = kc * 32;
      v8u ah = load_frag_a(ahip, kb, half);
      v8u al = load_frag_a(alop, kb, half);
      v8u bh = load_frag_b(bhip, kb, half);
      v8u bl = load_frag_b(blop, kb, half);
      C = wmma_bf16(ah, bh, C);   // hi*hi
      C = wmma_bf16(ah, bl, C);   // hi*lo
      C = wmma_bf16(al, bh, C);   // lo*hi
    }
    float sqn = sq[n];
    #pragma unroll
    for (int r = 0; r < 8; ++r) {
      int M = r + 8 * half;
      // squared distance: sqrt is monotone, defer it to the merge kernel
      sTile[w][M][nn] = sSq[rowGroup * 16 + M] + sqn - 2.f * C[r];
    }
    if (lane < 16) {                   // lane owns row `lane` of its group
      float* Lp = &sLd[w][lane][0];
      int*   Ip = &sLi[w][lane][0];
      #pragma unroll 1
      for (int t = 0; t < 16; ++t) {
        float d = sTile[w][lane][t];
        int j = colBase + t;
        if (d < rowmax && j != gi) {
          int ms = 0; float mv = Lp[0];
          #pragma unroll
          for (int s2 = 1; s2 < 16; ++s2)
            if (Lp[s2] > mv) { mv = Lp[s2]; ms = s2; }
          Lp[ms] = d; Ip[ms] = j;
          float nmax = Lp[0];
          #pragma unroll
          for (int s2 = 1; s2 < 16; ++s2) nmax = fmaxf(nmax, Lp[s2]);
          rowmax = nmax;
        }
      }
    }
  }
  for (int e = lane; e < 256; e += 32) {
    int rr = e >> 4, ee = e & 15;
    int grow = rowBase + rowGroup * 16 + rr;
    pld[(grow * PARTS + wsub) * 16 + ee] = sLd[w][rr][ee];
    pli[(grow * PARTS + wsub) * 16 + ee] = sLi[w][rr][ee];
  }
}

// ---------------------------------------------------------------------------
// Kernel 3: merge 4 partial top-16 lists/row -> 15 smallest, ascending,
// then convert d2 -> d (reference: where(d2>0, sqrt(max(d2,1e-24)), 0)).
// One wave per row (2 candidates per lane, wave-wide argmin 15 times).
// grid NPTS/8, block 256.
__global__ void lgr_merge(const float* __restrict__ pld, const int* __restrict__ pli,
                          float* __restrict__ kd, int* __restrict__ ki) {
  const int w = threadIdx.x >> 5, lane = threadIdx.x & 31;
  const int row = blockIdx.x * 8 + w;
  const float FINF = __builtin_huge_valf();
  float cd[2]; int ci[2];
  #pragma unroll
  for (int s = 0; s < 2; ++s) {
    int e = s * 32 + lane;
    cd[s] = pld[row * (PARTS * 16) + e];
    ci[s] = pli[row * (PARTS * 16) + e];
  }
  for (int sel = 0; sel < KNN; ++sel) {
    float bv = FINF; int bs = 0;
    #pragma unroll
    for (int s = 0; s < 2; ++s)
      if (cd[s] < bv) { bv = cd[s]; bs = s; }
    float mv = bv; int ml = lane;
    #pragma unroll
    for (int off = 16; off > 0; off >>= 1) {
      float ov = __shfl_xor(mv, off, 32);
      int   ol = __shfl_xor(ml, off, 32);
      if (ov < mv || (ov == mv && ol < ml)) { mv = ov; ml = ol; }
    }
    if (lane == ml) {
      float d = bv > 0.f ? sqrtf(fmaxf(bv, 1e-24f)) : 0.f;
      kd[row * KNN + sel] = d;
      ki[row * KNN + sel] = ci[bs];
      cd[bs] = FINF;
    }
  }
}

// ---------------------------------------------------------------------------
// Kernel 4: per-row distance-distribution loss partials + densities.
// grid NPTS/256, block 256.
__global__ void lgr_stats1(const float* __restrict__ kdC, const float* __restrict__ kdR,
                           float* __restrict__ densC, float* __restrict__ densR,
                           float* acc) {
  const int i = blockIdx.x * 256 + threadIdx.x;
  float cd[KNN], rd[KNN], mc = 0.f, mr = 0.f;
  #pragma unroll
  for (int t = 0; t < KNN; ++t) {
    cd[t] = kdC[i * KNN + t]; rd[t] = kdR[i * KNN + t];
    mc += cd[t]; mr += rd[t];
  }
  mc *= (1.f / KNN); mr *= (1.f / KNN);
  float ic = 1.f / (mc + EPSF), ir = 1.f / (mr + EPSF);
  float s = 0.f;
  #pragma unroll
  for (int t = 0; t < KNN; ++t) {
    float a = cd[t] * ic - rd[t] * ir;
    s += a * a;
  }
  atomicAdd(acc + 0, s);
  densC[i] = ic; densR[i] = ir;          // 1/(mean+eps)
  atomicAdd(acc + 3, ic);
  atomicAdd(acc + 4, ir);
}

// Kernel 5: density-preservation loss (needs global density means from acc).
__global__ void lgr_stats2(const float* __restrict__ densC, const float* __restrict__ densR,
                           float* acc) {
  const int i = blockIdx.x * 256 + threadIdx.x;
  float meanC = acc[3] * (1.f / NPTS), meanR = acc[4] * (1.f / NPTS);
  float a = densC[i] / (meanC + EPSF) - densR[i] / (meanR + EPSF);
  atomicAdd(acc + 1, a * a);
}

// ---------------------------------------------------------------------------
// Kernel 6: angular signature loss. One wave per row; per pass (ref, curr):
// stage 15 difference vectors as bf16 in LDS, compute 16x16 gram by WMMA,
// scale by 1/(|v_k||v_l|), accumulate sum of squared upper-triangle diffs.
// grid NPTS/ANGWAVES, block 32*ANGWAVES.
__global__ void lgr_angular(const float* __restrict__ emb, const float* __restrict__ ref,
                            const int* __restrict__ idxC, const int* __restrict__ idxR,
                            float* acc) {
  __shared__ unsigned short sD[ANGWAVES][16][DIMS + APAD];
  __shared__ float sRn[ANGWAVES][16];
  const int w = threadIdx.x >> 5, lane = threadIdx.x & 31;
  const int half = lane >> 4, nn = lane & 15;
  const int row = blockIdx.x * ANGWAVES + w;
  float cref[8];
  float s = 0.f;
  #pragma unroll 1
  for (int pass = 0; pass < 2; ++pass) {
    const float* E  = pass == 0 ? ref  : emb;
    const int* IDX  = pass == 0 ? idxR : idxC;
    #pragma unroll 1
    for (int r = 0; r < KNN; ++r) {
      int j = IDX[row * KNN + r];
      float p = 0.f;
      #pragma unroll
      for (int t = 0; t < 16; ++t) {
        int c = lane + 32 * t;
        float v = E[j * DIMS + c] - E[row * DIMS + c];
        p += v * v;
        sD[w][r][c] = f2bf_rne(v);
      }
      p = wave_sum(p);
      if (lane == 0) sRn[w][r] = 1.f / fmaxf(sqrtf(p), 1e-12f);
    }
    #pragma unroll
    for (int t = 0; t < 16; ++t) sD[w][15][lane + 32 * t] = 0;   // zero pad row
    if (lane == 0) sRn[w][15] = 0.f;

    const unsigned short* rowp = &sD[w][nn][0];
    v8f C = {};
    #pragma unroll 2
    for (int kc = 0; kc < DIMS / 32; ++kc) {
      int kb = kc * 32;
      v8u a = load_frag_a(rowp, kb, half);
      v8u b = load_frag_b(rowp, kb, half);
      C = wmma_bf16(a, b, C);
    }
    #pragma unroll
    for (int r = 0; r < 8; ++r) {
      int M = r + 8 * half;
      float cosv = C[r] * sRn[w][M] * sRn[w][nn];
      if (pass == 0) {
        cref[r] = cosv;
      } else if (M < nn && nn < KNN) {      // strict upper triangle of 15x15
        float dd = cosv - cref[r];
        s += dd * dd;
      }
    }
  }
  s = wave_sum(s);
  if (lane == 0) atomicAdd(acc + 2, s);
}

// ---------------------------------------------------------------------------
// Kernel 7: assemble the scalar loss.
__global__ void lgr_final(const float* acc, float* out) {
  out[0] = acc[0] * (1.f / (float(NPTS) * KNN))
         + 0.5f * acc[1] * (1.f / float(NPTS))
         + 0.5f * acc[2] * (1.f / (float(NPTS) * NPAIR));
}

// ---------------------------------------------------------------------------
extern "C" void kernel_launch(void* const* d_in, const int* in_sizes, int n_in,
                              void* d_out, int out_size, void* d_ws, size_t ws_size,
                              hipStream_t stream) {
  (void)in_sizes; (void)n_in; (void)out_size; (void)ws_size;
  const float* emb = (const float*)d_in[0];
  const float* ref = (const float*)d_in[1];
  float* out = (float*)d_out;

  size_t off = 0;
  auto carve = [&](size_t bytes) -> void* {
    void* p = (char*)d_ws + off;
    off += (bytes + 255) & ~(size_t)255;
    return p;
  };
  unsigned short* EhiE = (unsigned short*)carve((size_t)NPTS * DIMS * 2);
  unsigned short* EloE = (unsigned short*)carve((size_t)NPTS * DIMS * 2);
  unsigned short* EhiR = (unsigned short*)carve((size_t)NPTS * DIMS * 2);
  unsigned short* EloR = (unsigned short*)carve((size_t)NPTS * DIMS * 2);
  float* sqE  = (float*)carve((size_t)NPTS * 4);
  float* sqR  = (float*)carve((size_t)NPTS * 4);
  float* pld  = (float*)carve((size_t)NPTS * PARTS * 16 * 4);
  int*   pli  = (int*)  carve((size_t)NPTS * PARTS * 16 * 4);
  float* kdC  = (float*)carve((size_t)NPTS * KNN * 4);
  int*   kiC  = (int*)  carve((size_t)NPTS * KNN * 4);
  float* kdR  = (float*)carve((size_t)NPTS * KNN * 4);
  int*   kiR  = (int*)  carve((size_t)NPTS * KNN * 4);
  float* densC = (float*)carve((size_t)NPTS * 4);
  float* densR = (float*)carve((size_t)NPTS * 4);
  float* acc   = (float*)carve(64);

  lgr_init<<<1, 32, 0, stream>>>(acc);
  lgr_prep<<<dim3(NPTS / 8, 2), 256, 0, stream>>>(emb, ref, EhiE, EloE, sqE, EhiR, EloR, sqR);

  lgr_knn<<<NPTS / MBLK, 256, 0, stream>>>(EhiE, EloE, sqE, pld, pli);
  lgr_merge<<<NPTS / 8, 256, 0, stream>>>(pld, pli, kdC, kiC);

  lgr_knn<<<NPTS / MBLK, 256, 0, stream>>>(EhiR, EloR, sqR, pld, pli);
  lgr_merge<<<NPTS / 8, 256, 0, stream>>>(pld, pli, kdR, kiR);

  lgr_stats1<<<NPTS / 256, 256, 0, stream>>>(kdC, kdR, densC, densR, acc);
  lgr_stats2<<<NPTS / 256, 256, 0, stream>>>(densC, densR, acc);

  lgr_angular<<<NPTS / ANGWAVES, 32 * ANGWAVES, 0, stream>>>(emb, ref, kiC, kiR, acc);

  lgr_final<<<1, 1, 0, stream>>>(acc, out);
}
